// Int8Qwen2MLP_70111046140276
// MI455X (gfx1250) — 8-bit matrix operand order repaired and checked by reading, compiled, not run on hardware
//
#include <hip/hip_runtime.h>
#include <stdint.h>

#define T_TOKENS 2048
#define H_DIM    3584
#define I_DIM    18944
#define KSTEP    128

typedef int v8i  __attribute__((ext_vector_type(8)));
typedef unsigned int u32x4 __attribute__((ext_vector_type(4)));
typedef int i32x8 __attribute__((ext_vector_type(8)));
typedef int i32x4 __attribute__((ext_vector_type(4)));

__device__ __forceinline__ float wave_reduce_max(float v) {
#pragma unroll
    for (int off = 16; off > 0; off >>= 1)
        v = fmaxf(v, __shfl_xor(v, off, 32));
    return v;
}

__device__ __forceinline__ uint32_t lds_addr(const void* p) {
    return (uint32_t)(uintptr_t)p;
}

__device__ __forceinline__ void tdm_load_tile(uint32_t lds_byte_addr,
                                              const void* gbase,
                                              uint64_t byte_off,
                                              uint32_t row_stride) {
    const uint64_t ga = (uint64_t)(uintptr_t)gbase + byte_off;
    u32x4 g0;
    g0[0] = 1u;
    g0[1] = lds_byte_addr;
    g0[2] = (uint32_t)ga;
    g0[3] = (uint32_t)(ga >> 32) | (2u << 30);
    i32x8 g1;
    g1[0] = 0;
    g1[1] = 0;
    g1[2] = 0x4000;
    g1[3] = 0x4000 | (KSTEP << 16);
    g1[4] = 64;
    g1[5] = (int)row_stride;
    g1[6] = 0;
    g1[7] = 0;
    const i32x4 z4 = {0, 0, 0, 0};
    const i32x8 z8 = {0, 0, 0, 0, 0, 0, 0, 0};
    __builtin_amdgcn_tensor_load_to_lds(g0, g1, z4, z4, z8, 0);
}

__global__ __launch_bounds__(256)
void quant_x_kernel(const float* __restrict__ x,
                    int8_t* __restrict__ qx,
                    float* __restrict__ sx) {
    const int t   = blockIdx.x;
    const int tid = threadIdx.x;
    const int n4  = H_DIM / 4;

    const float4* xrow = (const float4*)(x + (size_t)t * H_DIM);
    float m = 0.0f;
    for (int i = tid; i < n4; i += 256) {
        float4 v = xrow[i];
        m = fmaxf(m, fmaxf(fmaxf(fabsf(v.x), fabsf(v.y)),
                           fmaxf(fabsf(v.z), fabsf(v.w))));
    }

    __shared__ float red[8];
    m = wave_reduce_max(m);
    const int wave = tid >> 5, lane = tid & 31;
    if (lane == 0) red[wave] = m;
    __syncthreads();
    if (wave == 0) {
        float v = (lane < 8) ? red[lane] : 0.0f;
        v = wave_reduce_max(v);
        if (lane == 0) {
            float s = fmaxf(v * (1.0f / 127.0f), 1e-8f);
            red[0] = s;
            sx[t] = s;
        }
    }
    __syncthreads();

    const float inv = 1.0f / red[0];
    char4* qrow = (char4*)(qx + (size_t)t * H_DIM);
    for (int i = tid; i < n4; i += 256) {
        float4 v = xrow[i];
        int a = (int)rintf(v.x * inv);
        int b = (int)rintf(v.y * inv);
        int c = (int)rintf(v.z * inv);
        int d = (int)rintf(v.w * inv);
        a = a > 127 ? 127 : (a < -127 ? -127 : a);
        b = b > 127 ? 127 : (b < -127 ? -127 : b);
        c = c > 127 ? 127 : (c < -127 ? -127 : c);
        d = d > 127 ? 127 : (d < -127 ? -127 : d);
        qrow[i] = make_char4((signed char)a, (signed char)b,
                             (signed char)c, (signed char)d);
    }
}

__device__ __forceinline__ v8i load_a_frag(const int8_t* lds, int row,
                                           int kk, int hi) {
    v8i a;
    const int8_t* p = lds + row * KSTEP + kk + 16 * hi;
#pragma unroll
    for (int c = 0; c < 4; ++c) {
        int2 d = *(const int2*)(p + 8 * (c & 1) + 32 * (c >> 1));
        a[2 * c]     = d.x;
        a[2 * c + 1] = d.y;
    }
    return a;
}

__device__ __forceinline__ v8i load_b_frag(const int8_t* lds, int col,
                                           int kk, int hi) {
    const int8_t* p = lds + col * KSTEP + kk + 16 * hi;
    int4 lo = *(const int4*)(p);
    int4 up = *(const int4*)(p + 32);
    v8i b;
    b[0] = lo.x; b[1] = lo.y; b[2] = lo.z; b[3] = lo.w;
    b[4] = up.x; b[5] = up.y; b[6] = up.z; b[7] = up.w;
    return b;
}

#define TILE_BYTES (64 * KSTEP)

__global__ __launch_bounds__(256)
void gateup_swiglu_kernel(const int8_t* __restrict__ qx,
                          const float* __restrict__ sx,
                          const int8_t* __restrict__ gate_w,
                          const int8_t* __restrict__ up_w,
                          const float* __restrict__ gws_p,
                          const float* __restrict__ uws_p,
                          const float* __restrict__ dis_p,
                          int8_t* __restrict__ qh) {
    __shared__ __align__(16) int8_t lds_a[2][TILE_BYTES];
    __shared__ __align__(16) int8_t lds_bg[2][TILE_BYTES];
    __shared__ __align__(16) int8_t lds_bu[2][TILE_BYTES];

    const int tid  = threadIdx.x;
    const int lane = tid & 31;
    const int wave = __builtin_amdgcn_readfirstlane(tid >> 5);
    const int m0   = blockIdx.y * 64;
    const int n0   = blockIdx.x * 64;
    const int wm   = wave & 1;
    const int wn   = wave >> 1;

    v8i acc_g0 = {}, acc_g1 = {};
    v8i acc_u0 = {}, acc_u1 = {};

    const uint64_t ga0 = (uint64_t)((size_t)m0 * H_DIM);
    const uint64_t gb0 = (uint64_t)((size_t)n0 * H_DIM);

    auto stage = [&](int buf, int k0) {
        if (wave == 0)
            tdm_load_tile(lds_addr(&lds_a[buf][0]),  qx,     ga0 + k0, H_DIM);
        else if (wave == 1)
            tdm_load_tile(lds_addr(&lds_bg[buf][0]), gate_w, gb0 + k0, H_DIM);
        else if (wave == 2)
            tdm_load_tile(lds_addr(&lds_bu[buf][0]), up_w,   gb0 + k0, H_DIM);
    };

    const int hi   = lane >> 4;
    const int mr0  = wm * 32 + (lane & 15);
    const int ncol = wn * 16 + (lane & 15);

    auto compute = [&](int buf) {
        const int8_t* A  = &lds_a[buf][0];
        const int8_t* Bg = &lds_bg[buf][0];
        const int8_t* Bu = &lds_bu[buf][0];
#pragma unroll
        for (int kk = 0; kk < KSTEP; kk += 64) {
            v8i a0 = load_a_frag(A, mr0,      kk, hi);
            v8i a1 = load_a_frag(A, mr0 + 16, kk, hi);
            v8i bg = load_b_frag(Bg, ncol,    kk, hi);
            acc_g0 = __builtin_amdgcn_wmma_i32_16x16x64_iu8(
                true, a0, true, bg, acc_g0, false, false);
            acc_g1 = __builtin_amdgcn_wmma_i32_16x16x64_iu8(
                true, a1, true, bg, acc_g1, false, false);
            v8i bu = load_b_frag(Bu, ncol,    kk, hi);
            acc_u0 = __builtin_amdgcn_wmma_i32_16x16x64_iu8(
                true, a0, true, bu, acc_u0, false, false);
            acc_u1 = __builtin_amdgcn_wmma_i32_16x16x64_iu8(
                true, a1, true, bu, acc_u1, false, false);
        }
    };

    stage(0, 0);
    int cur = 0;
    for (int k0 = 0; k0 + KSTEP < H_DIM; k0 += KSTEP) {
        stage(cur ^ 1, k0 + KSTEP);
        __builtin_amdgcn_s_wait_tensorcnt(1);
        __syncthreads();
        compute(cur);
        __syncthreads();
        cur ^= 1;
    }
    __builtin_amdgcn_s_wait_tensorcnt(0);
    __syncthreads();
    compute(cur);

    const float gws     = *gws_p;
    const float uws     = *uws_p;
    const float inv_dis = 1.0f / *dis_p;
    const int nout = n0 + wn * 16 + (lane & 15);
#pragma unroll
    for (int s = 0; s < 2; ++s) {
        const v8i& ag = s ? acc_g1 : acc_g0;
        const v8i& au = s ? acc_u1 : acc_u0;
        const int mbase = m0 + wm * 32 + s * 16 + 8 * hi;
#pragma unroll
        for (int i = 0; i < 8; ++i) {
            const int m = mbase + i;
            const float sc = sx[m];
            const float g  = (float)ag[i] * sc * gws;
            const float u  = (float)au[i] * sc * uws;
            const float hv = (g / (1.0f + __expf(-g))) * u;
            int q = (int)rintf(hv * inv_dis);
            q = q > 127 ? 127 : (q < -127 ? -127 : q);
            qh[(size_t)m * I_DIM + nout] = (int8_t)q;
        }
    }
}

__global__ __launch_bounds__(256)
void down_kernel(const int8_t* __restrict__ qh,
                 const int8_t* __restrict__ down_w,
                 const float* __restrict__ dws_p,
                 const float* __restrict__ dis_p,
                 float* __restrict__ out) {
    __shared__ __align__(16) int8_t lds_a[2][TILE_BYTES];
    __shared__ __align__(16) int8_t lds_b[2][TILE_BYTES];

    const int tid  = threadIdx.x;
    const int lane = tid & 31;
    const int wave = __builtin_amdgcn_readfirstlane(tid >> 5);
    const int m0   = blockIdx.y * 64;
    const int n0   = blockIdx.x * 64;
    const int wm   = wave & 1;
    const int wn   = wave >> 1;

    v8i acc0 = {}, acc1 = {};

    const uint64_t ga0 = (uint64_t)((size_t)m0 * I_DIM);
    const uint64_t gb0 = (uint64_t)((size_t)n0 * I_DIM);

    auto stage = [&](int buf, int k0) {
        if (wave == 0)
            tdm_load_tile(lds_addr(&lds_a[buf][0]), qh,     ga0 + k0, I_DIM);
        else if (wave == 1)
            tdm_load_tile(lds_addr(&lds_b[buf][0]), down_w, gb0 + k0, I_DIM);
    };

    const int hi   = lane >> 4;
    const int mr0  = wm * 32 + (lane & 15);
    const int ncol = wn * 16 + (lane & 15);

    auto compute = [&](int buf) {
        const int8_t* A = &lds_a[buf][0];
        const int8_t* B = &lds_b[buf][0];
#pragma unroll
        for (int kk = 0; kk < KSTEP; kk += 64) {
            v8i a0 = load_a_frag(A, mr0,      kk, hi);
            v8i a1 = load_a_frag(A, mr0 + 16, kk, hi);
            v8i b  = load_b_frag(B, ncol,     kk, hi);
            acc0 = __builtin_amdgcn_wmma_i32_16x16x64_iu8(
                true, a0, true, b, acc0, false, false);
            acc1 = __builtin_amdgcn_wmma_i32_16x16x64_iu8(
                true, a1, true, b, acc1, false, false);
        }
    };

    stage(0, 0);
    int cur = 0;
    for (int k0 = 0; k0 + KSTEP < I_DIM; k0 += KSTEP) {
        stage(cur ^ 1, k0 + KSTEP);
        __builtin_amdgcn_s_wait_tensorcnt(1);
        __syncthreads();
        compute(cur);
        __syncthreads();
        cur ^= 1;
    }
    __builtin_amdgcn_s_wait_tensorcnt(0);
    __syncthreads();
    compute(cur);

    const float scale = (*dis_p) * (*dws_p);
    const int nout = n0 + wn * 16 + (lane & 15);
#pragma unroll
    for (int s = 0; s < 2; ++s) {
        const v8i& a = s ? acc1 : acc0;
        const int mbase = m0 + wm * 32 + s * 16 + 8 * hi;
#pragma unroll
        for (int i = 0; i < 8; ++i) {
            const int m = mbase + i;
            out[(size_t)m * H_DIM + nout] = (float)a[i] * scale;
        }
    }
}

extern "C" void kernel_launch(void* const* d_in, const int* in_sizes, int n_in,
                              void* d_out, int out_size, void* d_ws, size_t ws_size,
                              hipStream_t stream) {
    const float*  x      = (const float*)d_in[0];
    const int8_t* gate_w = (const int8_t*)d_in[1];
    const int8_t* up_w   = (const int8_t*)d_in[2];
    const int8_t* down_w = (const int8_t*)d_in[3];
    const float*  gws    = (const float*)d_in[4];
    const float*  uws    = (const float*)d_in[5];
    const float*  dws    = (const float*)d_in[6];
    const float*  dis    = (const float*)d_in[7];
    float* out = (float*)d_out;

    uint8_t* ws = (uint8_t*)d_ws;
    const size_t qx_bytes = (size_t)T_TOKENS * H_DIM;
    const size_t sx_off   = (qx_bytes + 255) & ~(size_t)255;
    const size_t qh_off   = (sx_off + (size_t)T_TOKENS * 4 + 255) & ~(size_t)255;
    int8_t* qx = (int8_t*)ws;
    float*  sx = (float*)(ws + sx_off);
    int8_t* qh = (int8_t*)(ws + qh_off);

    quant_x_kernel<<<T_TOKENS, 256, 0, stream>>>(x, qx, sx);

    dim3 g2(I_DIM / 64, T_TOKENS / 64);
    gateup_swiglu_kernel<<<g2, 256, 0, stream>>>(qx, sx, gate_w, up_w,
                                                 gws, uws, dis, qh);

    dim3 g3(H_DIM / 64, T_TOKENS / 64);
    down_kernel<<<g3, 256, 0, stream>>>(qh, down_w, dws, dis, out);
}
